// DeepCluster_28217935134950
// MI455X (gfx1250) — matrix operand order repaired and checked by reading, not compiled, not run on hardware
//
#include <hip/hip_runtime.h>

typedef _Float16 v16h __attribute__((ext_vector_type(16)));
typedef _Float16 v8h  __attribute__((ext_vector_type(8)));
typedef float    v8f  __attribute__((ext_vector_type(8)));

#define MT 64
#define MTILES 4
#define THREADS 256

__device__ __forceinline__ v16h cat8(v8h a, v8h b) {
  return __builtin_shufflevector(a, b, 0,1,2,3,4,5,6,7,8,9,10,11,12,13,14,15);
}

// One fused layer: C[MT][NP] = A[MT][KP](LDS f16) * WT[NP][KP](global f16) (+bias, relu)
// WMMA 16x16x32 f16->f32. 8 waves each own n-tiles nt, nt+8, ...; B fragment reused
// across MTILES m-tiles.
template<int KP, int NP, int RELU, int SRCP, int DSTP, int F32OUT>
__device__ __forceinline__ void gemm_layer(
    const _Float16* __restrict__ src, void* __restrict__ dstv,
    const _Float16* __restrict__ wT, const float* __restrict__ bias,
    int wave, int lane)
{
  constexpr int NT = NP / 16;
  constexpr int KS = KP / 32;
  const int lm    = lane & 15;
  const int hi    = (lane & 16) ? 1 : 0;
  const int ahalf = hi ? 8  : 0;   // A frag: lanes 16-31 hold K+8 / K+24 halves
  const int bkoff = hi ? 8  : 0;   // B frag: lanes 16-31 hold K+8 / K+24 halves

  for (int nt = wave; nt < NT; nt += 8) {
    v8f acc[MTILES] = {};
    const _Float16* wp = wT + (size_t)(nt * 16 + lm) * KP + bkoff;
    #pragma unroll
    for (int k = 0; k < KS; ++k) {
      const v8h* bp = (const v8h*)(wp + k * 32);
      v16h bfrag = cat8(bp[0], bp[2]);
      #pragma unroll
      for (int m = 0; m < MTILES; ++m) {
        const _Float16* ap = src + (m * 16 + lm) * SRCP + k * 32 + ahalf;
        v16h afrag = cat8(*(const v8h*)ap, *(const v8h*)(ap + 16));
        acc[m] = __builtin_amdgcn_wmma_f32_16x16x32_f16(
            false, afrag, false, bfrag, (short)0, acc[m], false, false);
      }
    }
    const int col = nt * 16 + lm;
    if constexpr (F32OUT) {
      float* dst = (float*)dstv;
      #pragma unroll
      for (int m = 0; m < MTILES; ++m) {
        #pragma unroll
        for (int r = 0; r < 8; ++r) {
          int row = m * 16 + hi * 8 + r;   // C layout: lanes>=16 hold M+8
          dst[row * DSTP + col] = acc[m][r];
        }
      }
    } else {
      const float bv = bias[col];
      _Float16* dst = (_Float16*)dstv;
      #pragma unroll
      for (int m = 0; m < MTILES; ++m) {
        #pragma unroll
        for (int r = 0; r < 8; ++r) {
          float v = acc[m][r] + bv;
          if (RELU) v = v > 0.0f ? v : 0.0f;
          int row = m * 16 + hi * 8 + r;
          dst[row * DSTP + col] = (_Float16)v;
        }
      }
    }
  }
}

__global__ __launch_bounds__(THREADS, 1)
void dec_fused_kernel(
    const float* __restrict__ in,
    const _Float16* __restrict__ wt1, const _Float16* __restrict__ wt2,
    const _Float16* __restrict__ wt3, const _Float16* __restrict__ wt4,
    const _Float16* __restrict__ wt5, const _Float16* __restrict__ wt6,
    const _Float16* __restrict__ wt7,
    const float* __restrict__ bp1, const float* __restrict__ bp2,
    const float* __restrict__ bp3, const float* __restrict__ bp4,
    const float* __restrict__ bp5, const float* __restrict__ bp6,
    const float* __restrict__ bp7,
    const _Float16* __restrict__ cT,    // [80][224] f16 (center transposed, padded)
    const float* __restrict__ csq,      // [80] sum_k center[k][j]^2
    float* __restrict__ out)
{
  // ping-pong activation buffers; pitch = width + 8 halves (bank-conflict-free b128)
  __shared__ __align__(16) _Float16 bufA[MT * 520];
  __shared__ __align__(16) _Float16 bufB[MT * 520];
  __shared__ float fsq[MT];

  const int tid  = threadIdx.x;
  const int wave = tid >> 5;
  const int lane = tid & 31;
  const size_t rowBase = (size_t)blockIdx.x * MT;

  // Stage in: input f32 [MT][72] -> bufA f16 [MT][96] (pitch 104), zero-padded
  for (int i = tid; i < MT * 104; i += THREADS) {
    int r = i / 104, c = i - r * 104;
    float v = (c < 72) ? in[(rowBase + r) * 72 + c] : 0.0f;
    bufA[i] = (_Float16)v;
  }
  __syncthreads();

  gemm_layer< 96, 128, 0, 104, 136, 0>(bufA, bufB, wt1, bp1, wave, lane); __syncthreads();
  gemm_layer<128, 256, 1, 136, 264, 0>(bufB, bufA, wt2, bp2, wave, lane); __syncthreads();
  gemm_layer<256, 256, 0, 264, 264, 0>(bufA, bufB, wt3, bp3, wave, lane); __syncthreads();
  gemm_layer<256, 512, 1, 264, 520, 0>(bufB, bufA, wt4, bp4, wave, lane); __syncthreads();
  gemm_layer<512, 512, 0, 520, 520, 0>(bufA, bufB, wt5, bp5, wave, lane); __syncthreads();
  gemm_layer<512, 512, 0, 520, 520, 0>(bufB, bufA, wt6, bp6, wave, lane); __syncthreads();
  gemm_layer<512, 224, 0, 520, 232, 0>(bufA, bufB, wt7, bp7, wave, lane); __syncthreads();

  // |f|^2 per row (cols 200..223 are exact zeros from padded weights/bias)
  if (tid < MT) {
    float s = 0.0f;
    const _Float16* f = bufB + tid * 232;
    for (int j = 0; j < 224; ++j) { float v = (float)f[j]; s += v * v; }
    fsq[tid] = s;
  }
  __syncthreads();

  // cross = f @ center : [MT][80] f32 into bufA (pitch 88 floats)
  gemm_layer<224, 80, 0, 232, 88, 1>(bufB, bufA, cT, nullptr, wave, lane);
  __syncthreads();

  // q[n,j] = (1/(1+sq)) / rowsum   (ALPHA=1 -> exponent -1)
  if (tid < MT) {
    float* cr = (float*)bufA + tid * 88;
    const float fs = fsq[tid];
    float s = 0.0f;
    for (int j = 0; j < 72; ++j) {
      float sq  = fs - 2.0f * cr[j] + csq[j];
      float nom = 1.0f / (1.0f + sq);
      cr[j] = nom;
      s += nom;
    }
    const float inv = 1.0f / s;
    float* op = out + (rowBase + tid) * 72;
    for (int j = 0; j < 72; j += 4) {
      float4 q = make_float4(cr[j] * inv, cr[j+1] * inv, cr[j+2] * inv, cr[j+3] * inv);
      *(float4*)(op + j) = q;
    }
  }
}

// --- prep kernels: transpose+convert weights to f16 [NP][KP], pad biases, center norms ---
__global__ void wprep_kernel(const float* __restrict__ w, _Float16* __restrict__ wT,
                             int K, int N, int KP, int total) {
  int idx = blockIdx.x * blockDim.x + threadIdx.x;
  if (idx >= total) return;
  int j = idx / KP, k = idx - j * KP;
  float v = (j < N && k < K) ? w[(size_t)k * N + j] : 0.0f;
  wT[idx] = (_Float16)v;
}

__global__ void bprep_kernel(const float* __restrict__ b, float* __restrict__ bp,
                             int N, int NP) {
  int j = blockIdx.x * blockDim.x + threadIdx.x;
  if (j < NP) bp[j] = (j < N) ? b[j] : 0.0f;
}

__global__ void csq_kernel(const float* __restrict__ center, float* __restrict__ csq) {
  int j = threadIdx.x;
  if (j < 80) {
    float s = 0.0f;
    if (j < 72) {
      for (int k = 0; k < 200; ++k) { float v = center[k * 72 + j]; s += v * v; }
    }
    csq[j] = s;
  }
}

extern "C" void kernel_launch(void* const* d_in, const int* in_sizes, int n_in,
                              void* d_out, int out_size, void* d_ws, size_t ws_size,
                              hipStream_t stream) {
  const float* in = (const float*)d_in[0];
  const float* w[7]; const float* b[7];
  for (int i = 0; i < 7; ++i) {
    w[i] = (const float*)d_in[1 + 2 * i];
    b[i] = (const float*)d_in[2 + 2 * i];
  }
  const float* center = (const float*)d_in[15];

  static const int K [8] = { 72,128,256,256,512,512,512,200};
  static const int Nn[8] = {128,256,256,512,512,512,200, 72};
  static const int KP[8] = { 96,128,256,256,512,512,512,224};
  static const int NP[8] = {128,256,256,512,512,512,224, 80};

  char* ws = (char*)d_ws;
  size_t off = 0;
  auto carve = [&](size_t bytes) -> char* {
    char* p = ws + off;
    off = (off + bytes + 255) & ~(size_t)255;
    return p;
  };

  _Float16* wT[8];
  for (int i = 0; i < 8; ++i) wT[i] = (_Float16*)carve((size_t)KP[i] * NP[i] * 2);
  float* bp[7];
  for (int i = 0; i < 7; ++i) bp[i] = (float*)carve((size_t)NP[i] * 4);
  float* csq = (float*)carve(80 * 4);

  for (int i = 0; i < 8; ++i) {
    const float* src = (i < 7) ? w[i] : center;
    int total = KP[i] * NP[i];
    wprep_kernel<<<(total + 255) / 256, 256, 0, stream>>>(src, wT[i], K[i], Nn[i], KP[i], total);
  }
  for (int i = 0; i < 7; ++i)
    bprep_kernel<<<(NP[i] + 255) / 256, 256, 0, stream>>>(b[i], bp[i], Nn[i], NP[i]);
  csq_kernel<<<1, 128, 0, stream>>>(center, csq);

  dec_fused_kernel<<<262144 / MT, THREADS, 0, stream>>>(
      in, wT[0], wT[1], wT[2], wT[3], wT[4], wT[5], wT[6],
      bp[0], bp[1], bp[2], bp[3], bp[4], bp[5], bp[6],
      wT[7], csq, (float*)d_out);
}
